// BatchRecurrentAttention16Layer_v2_40501541601426
// MI455X (gfx1250) — matrix operand order repaired and checked by reading, not compiled, not run on hardware
//
#include <hip/hip_runtime.h>
#include <hip/hip_bf16.h>
#include <cstdint>

// Problem constants (match reference)
#define M_MOD 8
#define B_DIM 64
#define S_DIM 128
#define D_DIM 512
#define H_DIM 8
#define HD    64
#define FF_DIM 1024

typedef __bf16 bf16_t;
typedef bf16_t v16bf __attribute__((ext_vector_type(16)));
typedef bf16_t v8bf  __attribute__((ext_vector_type(8)));
typedef bf16_t v2bf  __attribute__((ext_vector_type(2)));
typedef float  v8f   __attribute__((ext_vector_type(8)));

// f32 -> bf16: prefer the packed-convert builtin; otherwise use the native
// fptrunc (backend emits the gfx1250 bf16 convert; avoids scalar RNE soup).
__device__ __forceinline__ bf16_t f32_to_bf16(float f) {
#if __has_builtin(__builtin_amdgcn_cvt_pk_bf16_f32)
  v2bf r = __builtin_amdgcn_cvt_pk_bf16_f32(f, 0.f);
  return r[0];
#else
  return (bf16_t)f;
#endif
}
__device__ __forceinline__ float bf16_to_f32(bf16_t x) { return (float)x; }

// pack two f32 -> {lo,hi} bf16 pair in one 32-bit word
__device__ __forceinline__ unsigned pack2_bf16(float a, float b) {
#if __has_builtin(__builtin_amdgcn_cvt_pk_bf16_f32)
  v2bf r = __builtin_amdgcn_cvt_pk_bf16_f32(a, b);
#else
  v2bf r;
  r[0] = (bf16_t)a;
  r[1] = (bf16_t)b;
#endif
  return __builtin_bit_cast(unsigned, r);
}

// ---------------------------------------------------------------------------
// Tensor Data Mover: stage a 64x32 bf16 tile (row-major, row stride =
// row_stride elements) into LDS with a 40-element padded row stride
// (pad_interval = 16 DWORDs stored -> pad_amount = 4 DWORDs inserted).
// Guarded: 5-arg builtin on clang-22 (ROCm 7.2), 6-arg on clang-23.
// ---------------------------------------------------------------------------
#if __has_builtin(__builtin_amdgcn_tensor_load_to_lds)
#define HAVE_TDM 1
typedef unsigned u32x4 __attribute__((ext_vector_type(4)));
typedef int      i32x4 __attribute__((ext_vector_type(4)));
typedef int      i32x8 __attribute__((ext_vector_type(8)));

__device__ __forceinline__ void tdm_load_tile_bf16(const bf16_t* gsrc,
                                                   unsigned lds_byte_off,
                                                   unsigned tensor_d0,
                                                   unsigned tensor_d1,
                                                   long long row_stride) {
  unsigned long long ga = (unsigned long long)(uintptr_t)gsrc;
  u32x4 g0;
  g0[0] = 1u;                                         // count=1 (valid), user mode
  g0[1] = lds_byte_off;                               // lds_addr
  g0[2] = (unsigned)(ga & 0xffffffffu);               // global_addr[31:0]
  g0[3] = (unsigned)((ga >> 32) & 0x01ffffffu) | (2u << 30);  // addr[56:32], type=2
  i32x8 g1;
  g1[0] = (1 << 17)      // data_size = 1 (2 bytes)
        | (1 << 20)      // pad_enable
        | (3 << 22)      // pad_interval: 16 DWORDs between pads
        | (3 << 25);     // pad_amount: 4 DWORDs
  g1[1] = (int)((tensor_d0 & 0xffffu) << 16);                 // tensor_dim0[15:0]
  g1[2] = (int)((tensor_d0 >> 16) & 0xffffu)
        | (int)((tensor_d1 & 0xffffu) << 16);                 // dim0 hi | dim1 lo
  g1[3] = (int)((tensor_d1 >> 16) & 0xffffu) | (32 << 16);    // dim1 hi | tile_dim0=32
  g1[4] = 64;                                                 // tile_dim1=64, tile_dim2=0
  g1[5] = (int)(row_stride & 0xffffffffLL);                   // dim0_stride[31:0]
  g1[6] = (int)((row_stride >> 32) & 0xffffLL);               // dim0_stride[47:32]
  g1[7] = 0;
  i32x4 gz = {0, 0, 0, 0};
#if defined(__clang_major__) && (__clang_major__ >= 23)
  i32x8 gz8 = {0, 0, 0, 0, 0, 0, 0, 0};
  __builtin_amdgcn_tensor_load_to_lds(g0, g1, gz, gz, gz8, 0);
#else
  __builtin_amdgcn_tensor_load_to_lds(g0, g1, gz, gz, 0);
#endif
}
#else
#define HAVE_TDM 0
#endif

// ---------------------------------------------------------------------------
// Batched GEMM: C[z] = act( A[z % batchModA] (64 x K) @ W[z] (K x N) + bias[z] )
// 256 threads = 8 wave32s; block tile 64x128, K-step 32, LDS double-buffered
// with register prefetch (one barrier per K-step). Each wave: one A fragment
// reused by 4 WMMAs. 40-element padded rows -> every fragment = 2 ds_load_b128.
// All B fragments are gathered before the 4 back-to-back WMMAs (single dscnt
// wait per K-step).  CBF: 1 = bf16 out, 0 = f32.  ACT: 0 none,1 relu,2 tanh∘relu
// ---------------------------------------------------------------------------
template <typename TA, int CBF, int ACT>
__global__ __launch_bounds__(256)
void wmma_gemm(const TA* __restrict__ A, long strideA, int batchModA,
               const float* __restrict__ W, long strideW,
               const float* __restrict__ bias, long strideBias,
               void* __restrict__ Cv, long strideC,
               int N, int Kdim)
{
  __shared__ bf16_t As[2][64][40];
  __shared__ bf16_t BsT[2][128][40];

  constexpr bool kIsBF16 = (sizeof(TA) == 2);
  constexpr bool kUseTdm = (HAVE_TDM != 0) && kIsBF16;

  const int z = blockIdx.z;
  const TA*    Ab    = A + (long)(z % batchModA) * strideA;
  const float* Wb    = W + (long)z * strideW;
  const float* biasb = bias + (long)z * strideBias;

  const int tile_m = blockIdx.y * 64;
  const int tile_n = blockIdx.x * 128;
  const int tid  = threadIdx.x;
  const int wave = tid >> 5;
  const int lane = tid & 31;
  const int wrow = (wave >> 1) * 16;   // 0,16,32,48
  const int wcol = (wave & 1) * 64;    // 0,64

  // Per-thread staging coordinates.
  const int a_row0 = tid >> 3,          a_c40 = (tid & 7) * 4;          // f32 A
  const int a_row1 = (tid + 256) >> 3,  a_c41 = ((tid + 256) & 7) * 4;
  const int ab_row = tid >> 2,          ab_c8 = (tid & 3) * 8;          // bf16 A
  const int b_kp0  = tid >> 5,          b_n40 = (tid & 31) * 4;         // B k-pairs
  const int b_kp1  = (tid + 256) >> 5,  b_n41 = ((tid + 256) & 31) * 4;

  float4 ra0{}, ra1{};
  v8bf   ra8{};
  float4 rb_lo0{}, rb_hi0{}, rb_lo1{}, rb_hi1{};

  auto load_tiles = [&](int k0) {
    if constexpr (!kIsBF16) {
      ra0 = *(const float4*)&Ab[(long)(tile_m + a_row0) * Kdim + k0 + a_c40];
      ra1 = *(const float4*)&Ab[(long)(tile_m + a_row1) * Kdim + k0 + a_c41];
    } else if constexpr (!kUseTdm) {
      ra8 = *(const v8bf*)&Ab[(long)(tile_m + ab_row) * Kdim + k0 + ab_c8];
    }
    rb_lo0 = *(const float4*)&Wb[(long)(k0 + 2 * b_kp0)     * N + tile_n + b_n40];
    rb_hi0 = *(const float4*)&Wb[(long)(k0 + 2 * b_kp0 + 1) * N + tile_n + b_n40];
    rb_lo1 = *(const float4*)&Wb[(long)(k0 + 2 * b_kp1)     * N + tile_n + b_n41];
    rb_hi1 = *(const float4*)&Wb[(long)(k0 + 2 * b_kp1 + 1) * N + tile_n + b_n41];
  };

  v8f acc[4] = {v8f{}, v8f{}, v8f{}, v8f{}};

  load_tiles(0);
  int buf = 0;
  for (int k0 = 0; k0 < Kdim; k0 += 32) {
    // ---- store staged tile (for this k0) into LDS[buf] ------------------
    if constexpr (kUseTdm) {
#if HAVE_TDM
      if (wave == 0) {
        tdm_load_tile_bf16((const bf16_t*)(Ab + (long)tile_m * Kdim + k0),
                           (unsigned)(uintptr_t)&As[buf][0][0],
                           (unsigned)Kdim, 64u, (long long)Kdim);
      }
#endif
    } else if constexpr (kIsBF16) {
      *(v8bf*)&As[buf][ab_row][ab_c8] = ra8;
    } else {
      uint2 pA;
      pA.x = pack2_bf16(ra0.x, ra0.y); pA.y = pack2_bf16(ra0.z, ra0.w);
      *(uint2*)&As[buf][a_row0][a_c40] = pA;
      pA.x = pack2_bf16(ra1.x, ra1.y); pA.y = pack2_bf16(ra1.z, ra1.w);
      *(uint2*)&As[buf][a_row1][a_c41] = pA;
    }
    // B: transposed, (k,k+1) packed per 32-bit LDS store
    *(unsigned*)&BsT[buf][b_n40 + 0][2 * b_kp0] = pack2_bf16(rb_lo0.x, rb_hi0.x);
    *(unsigned*)&BsT[buf][b_n40 + 1][2 * b_kp0] = pack2_bf16(rb_lo0.y, rb_hi0.y);
    *(unsigned*)&BsT[buf][b_n40 + 2][2 * b_kp0] = pack2_bf16(rb_lo0.z, rb_hi0.z);
    *(unsigned*)&BsT[buf][b_n40 + 3][2 * b_kp0] = pack2_bf16(rb_lo0.w, rb_hi0.w);
    *(unsigned*)&BsT[buf][b_n41 + 0][2 * b_kp1] = pack2_bf16(rb_lo1.x, rb_hi1.x);
    *(unsigned*)&BsT[buf][b_n41 + 1][2 * b_kp1] = pack2_bf16(rb_lo1.y, rb_hi1.y);
    *(unsigned*)&BsT[buf][b_n41 + 2][2 * b_kp1] = pack2_bf16(rb_lo1.z, rb_hi1.z);
    *(unsigned*)&BsT[buf][b_n41 + 3][2 * b_kp1] = pack2_bf16(rb_lo1.w, rb_hi1.w);

    if constexpr (kUseTdm) {
#if HAVE_TDM && __has_builtin(__builtin_amdgcn_s_wait_tensorcnt)
      if (wave == 0) __builtin_amdgcn_s_wait_tensorcnt(0);
#endif
    }
    __syncthreads();

    // ---- issue global loads for the next tile (overlaps the WMMAs) ------
    if (k0 + 32 < Kdim) load_tiles(k0 + 32);

    // ---- fragments: each = two contiguous ds_load_b128 ------------------
    v16bf af;
    {
      const int row = wrow + (lane & 15);
      const int kb  = (lane >> 4) * 8;         // byte offsets 0 / 16: aligned
      v8bf alo = *(const v8bf*)&As[buf][row][kb];
      v8bf ahi = *(const v8bf*)&As[buf][row][16 + kb];
      af = __builtin_shufflevector(alo, ahi, 0, 1, 2, 3, 4, 5, 6, 7,
                                   8, 9, 10, 11, 12, 13, 14, 15);
    }
    // gather all B fragments first, then fire the 4 WMMAs back-to-back
    v16bf bfrag[4];
#pragma unroll
    for (int sub = 0; sub < 4; ++sub) {
      const int n0 = wcol + sub * 16 + (lane & 15);
      const int kb = (lane >> 4) * 8;          // byte offsets 0 / 16: aligned
      v8bf blo = *(const v8bf*)&BsT[buf][n0][kb];
      v8bf bhi = *(const v8bf*)&BsT[buf][n0][kb + 16];
      bfrag[sub] = __builtin_shufflevector(blo, bhi, 0, 1, 2, 3, 4, 5, 6, 7,
                                           8, 9, 10, 11, 12, 13, 14, 15);
    }
#pragma unroll
    for (int sub = 0; sub < 4; ++sub) {
      acc[sub] = __builtin_amdgcn_wmma_f32_16x16x32_bf16(
          false, af, false, bfrag[sub], (short)0, acc[sub], false, false);
    }
    buf ^= 1;
    // NOTE: no second barrier needed — next iteration stores to the other
    // buffer, and the per-iteration barrier orders reuse two steps later.
  }

  // ---- epilogue: bias + activation + store ------------------------------
  float*  Cf = (float*)Cv  + (long)z * strideC;
  bf16_t* Cb = (bf16_t*)Cv + (long)z * strideC;
#pragma unroll
  for (int sub = 0; sub < 4; ++sub) {
    const int col = tile_n + wcol + sub * 16 + (lane & 15);
    const float bvv = biasb[col];
#pragma unroll
    for (int e = 0; e < 8; ++e) {
      int row = tile_m + wrow + e + ((lane >> 4) << 3);
      float v = acc[sub][e] + bvv;
      if constexpr (ACT >= 1) v = fmaxf(v, 0.f);
      if constexpr (ACT == 2) v = tanhf(v);
      long i0 = (long)row * N + col;
      if constexpr (CBF) Cb[i0] = f32_to_bf16(v);
      else               Cf[i0] = v;
    }
  }
}

// ---------------------------------------------------------------------------
// Attention: one block per (m,b,h); query length 1 -> VALU (0.13 GF total).
// ---------------------------------------------------------------------------
__global__ __launch_bounds__(128)
void attention_kernel(const float* __restrict__ q,     // [M,B,D] f32
                      const bf16_t* __restrict__ Kp,   // [M,S*B,D] bf16
                      const bf16_t* __restrict__ Vp,   // [M,S*B,D] bf16
                      float* __restrict__ ao)          // [M,B,D] f32
{
  __shared__ float w[S_DIM];
  __shared__ float red[S_DIM];
  int idx = blockIdx.x;
  int h = idx % H_DIM; idx /= H_DIM;
  int b = idx % B_DIM; idx /= B_DIM;
  int m = idx;
  int t = threadIdx.x;

  const float*  qv   = q  + ((long)(m * B_DIM + b)) * D_DIM + h * HD;
  const bf16_t* krow = Kp + ((long)(m * S_DIM + t) * B_DIM + b) * D_DIM + h * HD;
  float s = 0.f;
#pragma unroll
  for (int c = 0; c < 8; ++c) {              // 8 x 16B loads of K
    v8bf kv = *(const v8bf*)&krow[c * 8];
#pragma unroll
    for (int j = 0; j < 8; ++j) s += qv[c * 8 + j] * bf16_to_f32(kv[j]);
  }
  s *= 0.125f;                               // 1/sqrt(64)
  w[t] = s; red[t] = s;
  __syncthreads();
  for (int off = 64; off > 0; off >>= 1) {
    if (t < off) red[t] = fmaxf(red[t], red[t + off]);
    __syncthreads();
  }
  float mx = red[0];
  __syncthreads();
  float e = __expf(w[t] - mx);
  w[t] = e; red[t] = e;
  __syncthreads();
  for (int off = 64; off > 0; off >>= 1) {
    if (t < off) red[t] += red[t + off];
    __syncthreads();
  }
  float inv = 1.f / red[0];
  __syncthreads();
  w[t] *= inv;
  __syncthreads();

  if (t < HD) {
    const bf16_t* vbase = Vp + ((long)m * S_DIM * B_DIM + b) * D_DIM + h * HD + t;
    float acc = 0.f;
    for (int si = 0; si < S_DIM; ++si)
      acc += w[si] * bf16_to_f32(vbase[(long)si * B_DIM * D_DIM]);
    ao[((long)(m * B_DIM + b)) * D_DIM + h * HD + t] = acc;
  }
}

// x = relu(concat(attn_out, prev_state)) -> bf16 [M,B,2D]
__global__ void build_x_kernel(const float* __restrict__ attn,
                               const float* __restrict__ prev_state,
                               bf16_t* __restrict__ x, int n)
{
  int i = blockIdx.x * blockDim.x + threadIdx.x;
  if (i >= n) return;
  int c = i % (2 * D_DIM);
  long mb = i / (2 * D_DIM);
  float v = (c < D_DIM) ? attn[mb * D_DIM + c]
                        : prev_state[mb * D_DIM + (c - D_DIM)];
  x[i] = f32_to_bf16(fmaxf(v, 0.f));
}

// g = sigmoid(hg . Wg2 + bg2) — one block per (g*M+m, b), FF=1024 reduction
__global__ __launch_bounds__(128)
void gate_kernel(const bf16_t* __restrict__ hg, const float* __restrict__ Wg2,
                 const float* __restrict__ bg2, float* __restrict__ g)
{
  __shared__ float red[128];
  int blk = blockIdx.x;          // = gm * B + b
  int b  = blk % B_DIM;
  int gm = blk / B_DIM;
  const bf16_t* hrow = hg + ((long)gm * B_DIM + b) * FF_DIM;
  const float*  wv   = Wg2 + (long)gm * FF_DIM;
  v8bf hv = *(const v8bf*)&hrow[threadIdx.x * 8];
  float4 w0 = *(const float4*)&wv[threadIdx.x * 8];
  float4 w1 = *(const float4*)&wv[threadIdx.x * 8 + 4];
  float acc = bf16_to_f32(hv[0]) * w0.x + bf16_to_f32(hv[1]) * w0.y +
              bf16_to_f32(hv[2]) * w0.z + bf16_to_f32(hv[3]) * w0.w +
              bf16_to_f32(hv[4]) * w1.x + bf16_to_f32(hv[5]) * w1.y +
              bf16_to_f32(hv[6]) * w1.z + bf16_to_f32(hv[7]) * w1.w;
  red[threadIdx.x] = acc;
  __syncthreads();
  for (int off = 64; off > 0; off >>= 1) {
    if (threadIdx.x < off) red[threadIdx.x] += red[threadIdx.x + off];
    __syncthreads();
  }
  if (threadIdx.x == 0) {
    float zz = red[0] + bg2[gm];
    g[blk] = 1.f / (1.f + __expf(-zz));
  }
}

// d_out[o,m,b,d] in order (state,query,key,value) = new[3],new[0],new[1],new[2]
__global__ void combine_kernel(const float* __restrict__ outv, const float* __restrict__ gv,
                               const float* __restrict__ pq, const float* __restrict__ pk,
                               const float* __restrict__ pv, const float* __restrict__ ps,
                               float* __restrict__ dst, int n)
{
  int i = blockIdx.x * blockDim.x + threadIdx.x;
  if (i >= n) return;
  int d = i % D_DIM;
  int b = (i / D_DIM) % B_DIM;
  int m = (i / (D_DIM * B_DIM)) % M_MOD;
  int o = i / (D_DIM * B_DIM * M_MOD);
  int gidx = (o == 0) ? 3 : (o - 1);
  float g  = gv[(gidx * M_MOD + m) * B_DIM + b];
  float ov = outv[(((long)gidx * M_MOD + m) * B_DIM + b) * D_DIM + d];
  const float* prev = (gidx == 0) ? pq : (gidx == 1) ? pk : (gidx == 2) ? pv : ps;
  float pr = prev[((long)m * B_DIM + b) * D_DIM + d];
  dst[i] = g * ov + (1.f - g) * pr;
}

extern "C" void kernel_launch(void* const* d_in, const int* in_sizes, int n_in,
                              void* d_out, int out_size, void* d_ws, size_t ws_size,
                              hipStream_t stream)
{
  (void)in_sizes; (void)n_in; (void)out_size; (void)ws_size;

  const float* prev_state = (const float*)d_in[0];
  const float* prev_query = (const float*)d_in[1];
  const float* prev_key   = (const float*)d_in[2];
  const float* prev_value = (const float*)d_in[3];
  const float* key_in     = (const float*)d_in[4];
  const float* value_in   = (const float*)d_in[5];
  const float* Wq = (const float*)d_in[6];  const float* bq = (const float*)d_in[7];
  const float* Wk = (const float*)d_in[8];  const float* bk = (const float*)d_in[9];
  const float* Wv = (const float*)d_in[10]; const float* bv = (const float*)d_in[11];
  const float* Wo = (const float*)d_in[12]; const float* bo = (const float*)d_in[13];
  const float* W1 = (const float*)d_in[14]; const float* b1 = (const float*)d_in[15];
  const float* W2 = (const float*)d_in[16]; const float* b2 = (const float*)d_in[17];
  const float* Wg1 = (const float*)d_in[18]; const float* bg1 = (const float*)d_in[19];
  const float* Wg2 = (const float*)d_in[20]; const float* bg2 = (const float*)d_in[21];

  // Workspace carve-out (~144 MiB), 256B-aligned chunks.
  char* wp = (char*)d_ws;
  auto carve = [&](size_t bytes) {
    char* p = wp; wp += (bytes + 255) & ~(size_t)255; return p;
  };
  float*  q_f32    = (float*) carve(sizeof(float)  * (size_t)M_MOD * B_DIM * D_DIM);
  float*  ao_f32   = (float*) carve(sizeof(float)  * (size_t)M_MOD * B_DIM * D_DIM);
  float*  attn_f32 = (float*) carve(sizeof(float)  * (size_t)M_MOD * B_DIM * D_DIM);
  bf16_t* K_bf     = (bf16_t*)carve(sizeof(bf16_t) * (size_t)M_MOD * S_DIM * B_DIM * D_DIM);
  bf16_t* V_bf     = (bf16_t*)carve(sizeof(bf16_t) * (size_t)M_MOD * S_DIM * B_DIM * D_DIM);
  bf16_t* x_bf     = (bf16_t*)carve(sizeof(bf16_t) * (size_t)M_MOD * B_DIM * 2 * D_DIM);
  bf16_t* h_bf     = (bf16_t*)carve(sizeof(bf16_t) * (size_t)4 * M_MOD * B_DIM * FF_DIM);
  bf16_t* hg_bf    = (bf16_t*)carve(sizeof(bf16_t) * (size_t)4 * M_MOD * B_DIM * FF_DIM);
  float*  out_f32  = (float*) carve(sizeof(float)  * (size_t)4 * M_MOD * B_DIM * D_DIM);
  float*  g_f32    = (float*) carve(sizeof(float)  * (size_t)4 * M_MOD * B_DIM);

  dim3 blk(256);

  // K/V projections: [S*B,512] @ Wk/Wv[m] -> bf16 [M, S*B, 512]
  wmma_gemm<float, 1, 0><<<dim3(D_DIM / 128, (S_DIM * B_DIM) / 64, M_MOD), blk, 0, stream>>>(
      key_in, 0L, 1, Wk, (long)D_DIM * D_DIM, bk, (long)D_DIM,
      K_bf, (long)S_DIM * B_DIM * D_DIM, D_DIM, D_DIM);
  wmma_gemm<float, 1, 0><<<dim3(D_DIM / 128, (S_DIM * B_DIM) / 64, M_MOD), blk, 0, stream>>>(
      value_in, 0L, 1, Wv, (long)D_DIM * D_DIM, bv, (long)D_DIM,
      V_bf, (long)S_DIM * B_DIM * D_DIM, D_DIM, D_DIM);

  // q projection: [64,512] @ Wq[m] -> f32 [M,B,512]
  wmma_gemm<float, 0, 0><<<dim3(D_DIM / 128, B_DIM / 64, M_MOD), blk, 0, stream>>>(
      prev_query, (long)B_DIM * D_DIM, M_MOD, Wq, (long)D_DIM * D_DIM, bq, (long)D_DIM,
      q_f32, (long)B_DIM * D_DIM, D_DIM, D_DIM);

  // Multi-head attention (query length 1 per module)
  attention_kernel<<<M_MOD * B_DIM * H_DIM, 128, 0, stream>>>(q_f32, K_bf, V_bf, ao_f32);

  // Output projection: ao @ Wo + bo -> f32 [M,B,512]
  wmma_gemm<float, 0, 0><<<dim3(D_DIM / 128, B_DIM / 64, M_MOD), blk, 0, stream>>>(
      ao_f32, (long)B_DIM * D_DIM, M_MOD, Wo, (long)D_DIM * D_DIM, bo, (long)D_DIM,
      attn_f32, (long)B_DIM * D_DIM, D_DIM, D_DIM);

  // x = relu(concat(attn_out, prev_state)) -> bf16 [M,B,1024]
  {
    int n = M_MOD * B_DIM * 2 * D_DIM;
    build_x_kernel<<<(n + 255) / 256, 256, 0, stream>>>(attn_f32, prev_state, x_bf, n);
  }

  // h  = relu(x @ W1  + b1 ) -> bf16 [4,M,64,1024]   (z = g*M+m; A depends on m only)
  wmma_gemm<bf16_t, 1, 1><<<dim3(FF_DIM / 128, B_DIM / 64, 4 * M_MOD), blk, 0, stream>>>(
      x_bf, (long)B_DIM * 2 * D_DIM, M_MOD, W1, (long)2 * D_DIM * FF_DIM, b1, (long)FF_DIM,
      h_bf, (long)B_DIM * FF_DIM, FF_DIM, 2 * D_DIM);
  // hg = relu(x @ Wg1 + bg1) -> bf16 [4,M,64,1024]
  wmma_gemm<bf16_t, 1, 1><<<dim3(FF_DIM / 128, B_DIM / 64, 4 * M_MOD), blk, 0, stream>>>(
      x_bf, (long)B_DIM * 2 * D_DIM, M_MOD, Wg1, (long)2 * D_DIM * FF_DIM, bg1, (long)FF_DIM,
      hg_bf, (long)B_DIM * FF_DIM, FF_DIM, 2 * D_DIM);

  // out = tanh(relu(h @ W2 + b2)) -> f32 [4,M,64,512]
  wmma_gemm<bf16_t, 0, 2><<<dim3(D_DIM / 128, B_DIM / 64, 4 * M_MOD), blk, 0, stream>>>(
      h_bf, (long)B_DIM * FF_DIM, 4 * M_MOD, W2, (long)FF_DIM * D_DIM, b2, (long)D_DIM,
      out_f32, (long)B_DIM * D_DIM, D_DIM, FF_DIM);

  // scalar gates
  gate_kernel<<<4 * M_MOD * B_DIM, 128, 0, stream>>>(hg_bf, Wg2, bg2, g_f32);

  // gated blend into d_out, order (state, query, key, value)
  {
    int n = 4 * M_MOD * B_DIM * D_DIM;
    combine_kernel<<<(n + 255) / 256, 256, 0, stream>>>(
        out_f32, g_f32, prev_query, prev_key, prev_value, prev_state,
        (float*)d_out, n);
  }
}
